// WGCN_38912403701763
// MI455X (gfx1250) — matrix operand order repaired and checked by reading, not compiled, not run on hardware
//
#include <hip/hip_runtime.h>
#include <math.h>

#define NENT 100000
#define NEDGE 640000
#define DD 128

typedef __attribute__((ext_vector_type(16))) __bf16 v16bf;
typedef __attribute__((ext_vector_type(8)))  float  v8f;

// Fast, sign-symmetric tanh via hardware v_exp_f32. Accurate well past the
// ~1e-2 relative error already introduced by bf16 WMMA inputs.
__device__ __forceinline__ float fast_tanh(float x) {
    float ax = fabsf(x);
    float e  = __expf(2.0f * ax);
    float t  = 1.0f - 2.0f / (e + 1.0f);
    return copysignf(t, x);
}

// ---------------- utility kernels ----------------

__global__ void init_scalars(float* scal) {
    scal[0] = 0.f;                       // rel_term
    ((unsigned int*)scal)[1] = 0u;       // ordered-int max
    scal[2] = 0.f;                       // exp-sum
    scal[3] = 0.f;                       // max (float)
    scal[4] = 0.f;                       // 1/sum
}

__global__ void copy_f32(float* __restrict__ dst, const float* __restrict__ src, int n4) {
    int i = blockIdx.x * blockDim.x + threadIdx.x;
    if (i < n4) ((float4*)dst)[i] = ((const float4*)src)[i];
}

__global__ void zero_f32(float* __restrict__ dst, int n4) {
    int i = blockIdx.x * blockDim.x + threadIdx.x;
    if (i < n4) ((float4*)dst)[i] = make_float4(0.f, 0.f, 0.f, 0.f);
}

// ---------------- edge message scatter:  agg[head] += relw[rel] * x[tail] ----------------
// 32 threads per edge, 4 floats per thread.
__global__ __launch_bounds__(256)
void scatter_msg(float* __restrict__ agg, const float* __restrict__ x,
                 const int* __restrict__ heads, const int* __restrict__ rels,
                 const int* __restrict__ tails, const float* __restrict__ relw) {
    int idx  = blockIdx.x * blockDim.x + threadIdx.x;
    int edge = idx >> 5;
    if (edge >= NEDGE) return;
    int c = (idx & 31) * 4;
    int h = heads[edge], r = rels[edge], t = tails[edge];
    float w = relw[r];
    float4 v = *(const float4*)(x + (long)t * DD + c);
    float* o = agg + (long)h * DD + c;
    atomicAdd(o + 0, w * v.x);
    atomicAdd(o + 1, w * v.y);
    atomicAdd(o + 2, w * v.z);
    atomicAdd(o + 3, w * v.w);
}

// ---------------- pack 128x128 fp32 weight into WMMA B-fragment order (bf16) ----------------
// Layout index: (((ks*8 + nt)*32 + lane)*16 + j)
// B element for lane-group g = lane>>4 : K = ks*32 + 8*g + (j&7) + 16*(j>>3) ; N = nt*16 + (lane&15)
__global__ void pack_w(__bf16* __restrict__ wpk, const float* __restrict__ W, int transpose) {
    int tid = blockIdx.x * blockDim.x + threadIdx.x;
    if (tid >= 4 * 8 * 32 * 16) return;
    int j    = tid & 15;
    int lane = (tid >> 4) & 31;
    int nt   = (tid >> 9) & 7;
    int ks   = tid >> 12;
    int g = lane >> 4;
    int n = nt * 16 + (lane & 15);
    int k = ks * 32 + g * 8 + (j & 7) + ((j >> 3) << 4);
    float v = transpose ? W[n * DD + k] : W[k * DD + n];
    wpk[tid] = (__bf16)v;
}

// ---------------- GEMM: Out[M,128] = (tanh?) (A[M,128] @ Wpacked) via v_wmma_f32_16x16x32_bf16 ----
// 8 waves/block; wave owns 16 rows x all 128 cols. Safe in-place (all A reads precede stores).
__global__ __launch_bounds__(256)
void gemm_wmma(const float* A, const __bf16* __restrict__ wpk, float* Out, int apply_tanh) {
    int waveId = threadIdx.x >> 5;
    int lane   = threadIdx.x & 31;
    int mrow   = blockIdx.x * 128 + waveId * 16;
    if (mrow >= NENT) return;            // uniform per wave (100000 % 16 == 0)
    int L = lane & 15;
    int g = lane >> 4;

    v8f acc[8];
#pragma unroll
    for (int nt = 0; nt < 8; nt++)
#pragma unroll
        for (int r = 0; r < 8; r++) acc[nt][r] = 0.f;

    const float* arow = A + (long)(mrow + L) * DD;
#pragma unroll
    for (int ks = 0; ks < 4; ks++) {
        // A fragment: lane-group g holds K = ks*32 + 8g + {0..7} and +16..+23
        int k0 = ks * 32 + 8 * g;
        float4 a0 = *(const float4*)(arow + k0);
        float4 a1 = *(const float4*)(arow + k0 + 4);
        float4 a2 = *(const float4*)(arow + k0 + 16);
        float4 a3 = *(const float4*)(arow + k0 + 20);
        float af[16] = {a0.x, a0.y, a0.z, a0.w, a1.x, a1.y, a1.z, a1.w,
                        a2.x, a2.y, a2.z, a2.w, a3.x, a3.y, a3.z, a3.w};
        v16bf av;
#pragma unroll
        for (int j = 0; j < 16; j++) av[j] = (__bf16)af[j];   // hardware f32->bf16 cvt
#pragma unroll
        for (int nt = 0; nt < 8; nt++) {
            v16bf b = *(const v16bf*)(wpk + (((ks * 8 + nt) * 32 + lane) << 4));
            acc[nt] = __builtin_amdgcn_wmma_f32_16x16x32_bf16(
                false, av, false, b, (short)0, acc[nt], false, false);
        }
    }
#pragma unroll
    for (int nt = 0; nt < 8; nt++) {
#pragma unroll
        for (int r = 0; r < 8; r++) {
            float v = acc[nt][r];
            if (apply_tanh) v = fast_tanh(v);
            Out[(long)(mrow + r + 8 * g) * DD + nt * 16 + L] = v;
        }
    }
}

// ---------------- per-entity dots: s1 = xh . u1, s3 = xh . u3 ----------------
__global__ __launch_bounds__(128)
void ent_dots(const float* __restrict__ xh, const float* __restrict__ u,
              float* __restrict__ s1, float* __restrict__ s3) {
    __shared__ float sh1[128], sh3[128];
    int ent = blockIdx.x;
    int t = threadIdx.x;
    float v = xh[(long)ent * DD + t];
    sh1[t] = v * u[t];
    sh3[t] = v * u[2 * DD + t];
    __syncthreads();
    for (int s = 64; s > 0; s >>= 1) {
        if (t < s) { sh1[t] += sh1[t + s]; sh3[t] += sh3[t + s]; }
        __syncthreads();
    }
    if (t == 0) { s1[ent] = sh1[0]; s3[ent] = sh3[0]; }
}

// ---------------- rel_term = dot(re_weight @ rsa, u2) ----------------
__global__ __launch_bounds__(128)
void rel_term_kernel(const float* __restrict__ reW, const float* __restrict__ rsa,
                     const float* __restrict__ u, float* __restrict__ scal) {
    __shared__ float sh[128];
    int t = threadIdx.x;
    float acc = 0.f;
    for (int j = 0; j < DD; j++) acc += reW[t * DD + j] * rsa[j];
    sh[t] = acc * u[DD + t];
    __syncthreads();
    for (int s = 64; s > 0; s >>= 1) {
        if (t < s) sh[t] += sh[t + s];
        __syncthreads();
    }
    if (t == 0) scal[0] = sh[0];
}

// ---------------- edge scores + global max (ordered-int atomicMax) ----------------
__global__ __launch_bounds__(256)
void edge_e_max(const int* __restrict__ heads, const int* __restrict__ tails,
                const float* __restrict__ s1, const float* __restrict__ s3,
                float* __restrict__ eBuf, float* __restrict__ scal) {
    __shared__ float sh[256];
    int i = blockIdx.x * blockDim.x + threadIdx.x;
    float e = -3.402823e38f;
    if (i < NEDGE) {
        float val = s1[heads[i]] + s3[tails[i]] + scal[0];
        e = val > 0.f ? val : 0.01f * val;   // leaky_relu, slope 0.01
        eBuf[i] = e;
    }
    sh[threadIdx.x] = e;
    __syncthreads();
    for (int s = 128; s > 0; s >>= 1) {
        if (threadIdx.x < s) sh[threadIdx.x] = fmaxf(sh[threadIdx.x], sh[threadIdx.x + s]);
        __syncthreads();
    }
    if (threadIdx.x == 0) {
        unsigned int b = __float_as_uint(sh[0]);
        unsigned int m = (b & 0x80000000u) ? ~b : (b | 0x80000000u);
        atomicMax(((unsigned int*)scal) + 1, m);
    }
}

__global__ __launch_bounds__(256)
void edge_expsum(const float* __restrict__ eBuf, float* __restrict__ scal) {
    __shared__ float sh[256];
    int i = blockIdx.x * blockDim.x + threadIdx.x;
    unsigned int m = ((const unsigned int*)scal)[1];
    float mx = (m & 0x80000000u) ? __uint_as_float(m ^ 0x80000000u) : __uint_as_float(~m);
    float s = 0.f;
    if (i < NEDGE) s = expf(eBuf[i] - mx);
    sh[threadIdx.x] = s;
    __syncthreads();
    for (int k = 128; k > 0; k >>= 1) {
        if (threadIdx.x < k) sh[threadIdx.x] += sh[threadIdx.x + k];
        __syncthreads();
    }
    if (threadIdx.x == 0) atomicAdd(&scal[2], sh[0]);
}

__global__ void finalize_softmax(float* scal) {
    unsigned int m = ((unsigned int*)scal)[1];
    float mx = (m & 0x80000000u) ? __uint_as_float(m ^ 0x80000000u) : __uint_as_float(~m);
    scal[3] = mx;
    scal[4] = 1.0f / scal[2];
}

// ---------------- out[head] += alpha_e * x3[tail] ----------------
__global__ __launch_bounds__(256)
void scatter_out(float* __restrict__ out, const float* __restrict__ x3,
                 const int* __restrict__ heads, const int* __restrict__ tails,
                 const float* __restrict__ eBuf, const float* __restrict__ scal) {
    int idx  = blockIdx.x * blockDim.x + threadIdx.x;
    int edge = idx >> 5;
    if (edge >= NEDGE) return;
    int c = (idx & 31) * 4;
    float alpha = expf(eBuf[edge] - scal[3]) * scal[4];
    int h = heads[edge], t = tails[edge];
    float4 v = *(const float4*)(x3 + (long)t * DD + c);
    float* o = out + (long)h * DD + c;
    atomicAdd(o + 0, alpha * v.x);
    atomicAdd(o + 1, alpha * v.y);
    atomicAdd(o + 2, alpha * v.z);
    atomicAdd(o + 3, alpha * v.w);
}

// ---------------- host launcher ----------------

extern "C" void kernel_launch(void* const* d_in, const int* in_sizes, int n_in,
                              void* d_out, int out_size, void* d_ws, size_t ws_size,
                              hipStream_t stream) {
    const int*   heads = (const int*)d_in[0];
    const int*   rels  = (const int*)d_in[1];
    const int*   tails = (const int*)d_in[2];
    const float* em    = (const float*)d_in[3];
    const float* W1    = (const float*)d_in[4];
    const float* W2    = (const float*)d_in[5];
    const float* W3    = (const float*)d_in[6];
    const float* relw  = (const float*)d_in[7];
    const float* u     = (const float*)d_in[8];
    const float* enw   = (const float*)d_in[9];
    const float* reW   = (const float*)d_in[10];
    const float* rsa   = (const float*)d_in[11];
    float*       out   = (float*)d_out;

    char* ws = (char*)d_ws;
    float*  bufA = (float*)(ws + 0);                 // 51,200,000 B  (x / agg ping buffer)
    float*  eBuf = (float*)(ws + 51200000);          //  2,560,000 B
    float*  s1   = (float*)(ws + 53760000);          //    400,000 B
    float*  s3   = (float*)(ws + 54160128);          //    400,000 B
    __bf16* wpk  = (__bf16*)(ws + 54560256);         //     32,768 B
    float*  scal = (float*)(ws + 54593024);          //         64 B

    const int n4 = NENT * DD / 4;                    // 3,200,000 float4s
    const dim3 cpB(256), cpG((n4 + 255) / 256);      // 12500 blocks
    const dim3 scB(256), scG(NEDGE * 32 / 256);      // 80000 blocks
    const dim3 gmB(256), gmG((NENT + 127) / 128);    // 782 blocks
    const dim3 pkB(256), pkG(64);

    init_scalars<<<1, 1, 0, stream>>>(scal);

    // Layer 1: agg = em + segsum(msg(em)) ; x1 = tanh(agg @ W1)   (in bufA)
    copy_f32<<<cpG, cpB, 0, stream>>>(bufA, em, n4);
    scatter_msg<<<scG, scB, 0, stream>>>(bufA, em, heads, rels, tails, relw);
    pack_w<<<pkG, pkB, 0, stream>>>(wpk, W1, 0);
    gemm_wmma<<<gmG, gmB, 0, stream>>>(bufA, wpk, bufA, 1);

    // Layer 2 (use d_out as ping-pong buffer): x2 in out
    copy_f32<<<cpG, cpB, 0, stream>>>(out, bufA, n4);
    scatter_msg<<<scG, scB, 0, stream>>>(out, bufA, heads, rels, tails, relw);
    pack_w<<<pkG, pkB, 0, stream>>>(wpk, W2, 0);
    gemm_wmma<<<gmG, gmB, 0, stream>>>(out, wpk, out, 1);

    // Layer 3: x3 in bufA
    copy_f32<<<cpG, cpB, 0, stream>>>(bufA, out, n4);
    scatter_msg<<<scG, scB, 0, stream>>>(bufA, out, heads, rels, tails, relw);
    pack_w<<<pkG, pkB, 0, stream>>>(wpk, W3, 0);
    gemm_wmma<<<gmG, gmB, 0, stream>>>(bufA, wpk, bufA, 1);

    // xh = x3 @ en_weight.T  -> out (scratch use of d_out)
    pack_w<<<pkG, pkB, 0, stream>>>(wpk, enw, 1);
    gemm_wmma<<<gmG, gmB, 0, stream>>>(bufA, wpk, out, 0);

    // attention scalars, edge scores, global softmax
    ent_dots<<<dim3(NENT), dim3(128), 0, stream>>>(out, u, s1, s3);
    rel_term_kernel<<<1, 128, 0, stream>>>(reW, rsa, u, scal);
    edge_e_max<<<dim3(NEDGE / 256), dim3(256), 0, stream>>>(heads, tails, s1, s3, eBuf, scal);
    edge_expsum<<<dim3(NEDGE / 256), dim3(256), 0, stream>>>(eBuf, scal);
    finalize_softmax<<<1, 1, 0, stream>>>(scal);

    // out = segsum(alpha * x3[tails], heads)
    zero_f32<<<cpG, cpB, 0, stream>>>(out, n4);
    scatter_out<<<scG, scB, 0, stream>>>(out, bufA, heads, tails, eBuf, scal);
}
